// GATAE_22814866276794
// MI455X (gfx1250) — matrix operand order repaired and checked by reading, not compiled, not run on hardware
//
#include <hip/hip_runtime.h>
#include <hip/hip_fp16.h>

typedef __attribute__((ext_vector_type(16))) _Float16 v16h;
typedef __attribute__((ext_vector_type(8)))  float    v8f;

#define N_NODES 50000
#define N_EDGES 800000
#define E_TOTAL (N_EDGES + N_NODES)
#define NEG_SLOPE 0.2f

// ---------------------------------------------------------------- helpers
__device__ __forceinline__ void atomicMaxF(float* addr, float val) {
  // classic mixed-sign float max via int atomics (init must be negative, is -1e30)
  if (val >= 0.0f) atomicMax((int*)addr, __float_as_int(val));
  else             atomicMin((unsigned int*)addr, __float_as_uint(val));
}

__device__ __forceinline__ void edge_sd(const long long* __restrict__ ei, int e,
                                        int& s, int& d) {
  if (e < N_EDGES) { s = (int)ei[e]; d = (int)ei[N_EDGES + e]; }
  else             { s = e - N_EDGES; d = s; }   // self loop
}

// ---------------------------------------------------------------- converts
__global__ void k_f32_to_f16(const float* __restrict__ src,
                             _Float16* __restrict__ dst, int n) {
  int i = blockIdx.x * blockDim.x + threadIdx.x;
  if (i < n) dst[i] = (_Float16)src[i];
}

// W is KxN row-major f32; write Wt as NxK row-major f16 (so k-pairs are contiguous)
__global__ void k_w_transpose_f16(const float* __restrict__ W,
                                  _Float16* __restrict__ Wt, int K, int N) {
  int i = blockIdx.x * blockDim.x + threadIdx.x;
  if (i < K * N) {
    int k = i / N, n = i % N;
    Wt[(size_t)n * K + k] = (_Float16)W[i];
  }
}

__global__ void k_zero_f32(float* __restrict__ p, int n) {
  int i = blockIdx.x * blockDim.x + threadIdx.x;
  if (i < n) p[i] = 0.0f;
}

// ---------------------------------------------------------------- WMMA GEMM
// C[M x N] = A[M x K] * B[K x N], A row-major f16, Bt = B^T (N x K row-major f16).
// One wave32 per 16 x (16*NW) output slab: one A fragment feeds NW WMMAs per
// K-step (register blocking along N). v_wmma_f32_16x16x32_f16 throughout.
template <int NW>
__global__ void k_wmma_gemm(const _Float16* __restrict__ A,
                            const _Float16* __restrict__ Bt,
                            float* __restrict__ C, int M, int N, int K) {
  const int lane    = threadIdx.x & 31;
  const int wave    = blockIdx.x * (blockDim.x >> 5) + (threadIdx.x >> 5);
  const int ngroups = N / (16 * NW);
  const int total   = (M >> 4) * ngroups;
  if (wave >= total) return;                 // wave-uniform: EXEC stays all-ones
  const int mt = wave / ngroups;
  const int ng = wave % ngroups;
  const bool lo  = (lane < 16);
  const int  l15 = lane & 15;
  const int  mrow = mt * 16 + l15;           // A fragment row for this lane

  v8f acc[NW];
#pragma unroll
  for (int t = 0; t < NW; ++t) acc[t] = v8f{};

  for (int k0 = 0; k0 < K; k0 += 32) {
    // ---- A fragment (shared by all NW tiles this k-step)
    v16h a;
    unsigned int* ap = (unsigned int*)&a;
    const unsigned int* arow = (const unsigned int*)(A + (size_t)mrow * K + k0);
#pragma unroll
    for (int j = 0; j < 8; ++j) {
      // 16-bit A 16x32 lane map: lanes 0-15 hold K 0-7/16-23, lanes 16-31 K 8-15/24-31
      int ka = (j < 4) ? (2 * j + (lo ? 0 : 8)) : (16 + 2 * (j - 4) + (lo ? 0 : 8));
      ap[j] = arow[ka >> 1];
    }
#pragma unroll
    for (int t = 0; t < NW; ++t) {
      const int ncol = (ng * NW + t) * 16 + l15;
      v16h b;
      unsigned int* bp = (unsigned int*)&b;
      const unsigned int* brow = (const unsigned int*)(Bt + (size_t)ncol * K + k0);
#pragma unroll
      for (int j = 0; j < 8; ++j) {
        // B 32x16: lanes 0-15 hold K 0-7/16-23, lanes 16-31 K 8-15/24-31 (2 per dword)
        int kb = (j < 4) ? (2 * j + (lo ? 0 : 8)) : (16 + 2 * (j - 4) + (lo ? 0 : 8));
        bp[j] = brow[kb >> 1];
      }
      acc[t] = __builtin_amdgcn_wmma_f32_16x16x32_f16(false, a, false, b,
                                                      (short)0, acc[t], false, false);
    }
  }
  // C/D layout: VGPR r -> row mt*16 + r (+8 for upper lanes), col = tile*16 + (lane&15)
  const int rbase = mt * 16 + (lo ? 0 : 8);
#pragma unroll
  for (int t = 0; t < NW; ++t) {
    const int cc = (ng * NW + t) * 16 + l15;
#pragma unroll
    for (int r = 0; r < 8; ++r)
      C[(size_t)(rbase + r) * N + cc] = acc[t][r];
  }
}

// ---------------------------------------------------------------- attention prep
// al_s[n,h] = sum_c h[n,h,c]*a_src[h,c] ; al_d likewise ; init emax/denom.
__global__ void k_att_init(const float* __restrict__ h,
                           const float* __restrict__ a_s,
                           const float* __restrict__ a_d,
                           float* __restrict__ als, float* __restrict__ ald,
                           float* __restrict__ emax, float* __restrict__ denom,
                           int H, int C) {
  int i = blockIdx.x * blockDim.x + threadIdx.x;
  if (i >= N_NODES * H) return;
  int n = i / H, hh = i % H;
  const float* hp  = h  + (size_t)n * H * C + (size_t)hh * C;
  const float* asp = a_s + hh * C;
  const float* adp = a_d + hh * C;
  float ss = 0.f, sd = 0.f;
  for (int c = 0; c < C; ++c) { float v = hp[c]; ss += v * asp[c]; sd += v * adp[c]; }
  als[i] = ss; ald[i] = sd; emax[i] = -1e30f; denom[i] = 0.f;
}

// ---------------------------------------------------------------- edge passes
__device__ __forceinline__ float edge_e(const float* als, const float* ald,
                                        int s, int d, int hh, int H) {
  float v = als[s * H + hh] + ald[d * H + hh];
  return v > 0.f ? v : NEG_SLOPE * v;
}

__global__ void k_edge_max(const long long* __restrict__ ei,
                           const float* __restrict__ als,
                           const float* __restrict__ ald,
                           float* __restrict__ emax, int H) {
  int i = blockIdx.x * blockDim.x + threadIdx.x;
  if (i >= E_TOTAL * H) return;
  int e = i / H, hh = i % H, s, d;
  edge_sd(ei, e, s, d);
  atomicMaxF(&emax[d * H + hh], edge_e(als, ald, s, d, hh, H));
}

__global__ void k_edge_denom(const long long* __restrict__ ei,
                             const float* __restrict__ als,
                             const float* __restrict__ ald,
                             const float* __restrict__ emax,
                             float* __restrict__ denom, int H) {
  int i = blockIdx.x * blockDim.x + threadIdx.x;
  if (i >= E_TOTAL * H) return;
  int e = i / H, hh = i % H, s, d;
  edge_sd(ei, e, s, d);
  float v = edge_e(als, ald, s, d, hh, H);
  atomicAdd(&denom[d * H + hh], __expf(v - emax[d * H + hh]));
}

// One wave per edge, lane = channel: coalesced h[src] reads and coalesced
// atomicAdd bursts into out[dst]. With C==32 the head index is wave-uniform
// per iteration, so the alpha computation scalarizes.
__global__ void k_edge_msg(const long long* __restrict__ ei,
                           const float* __restrict__ als,
                           const float* __restrict__ ald,
                           const float* __restrict__ emax,
                           const float* __restrict__ denom,
                           const float* __restrict__ h,
                           float* __restrict__ out, int H, int C) {
  const int lane = threadIdx.x & 31;
  const int wave = blockIdx.x * (blockDim.x >> 5) + (threadIdx.x >> 5);
  if (wave >= E_TOTAL) return;
  int s, d;
  edge_sd(ei, wave, s, d);
  const int HC = H * C;
  const float* hp = h   + (size_t)s * HC;
  float*       op = out + (size_t)d * HC;
  for (int j = lane; j < HC; j += 32) {
    int hh = j / C;
    int dh = d * H + hh;
    float v = als[s * H + hh] + ald[dh];
    v = v > 0.f ? v : NEG_SLOPE * v;
    float alpha = __expf(v - emax[dh]) / denom[dh];
    atomicAdd(&op[j], alpha * hp[j]);
  }
}

__global__ void k_bias_act(float* __restrict__ acc, const float* __restrict__ b,
                           float* __restrict__ out, int HC, int relu) {
  int i = blockIdx.x * blockDim.x + threadIdx.x;
  if (i >= N_NODES * HC) return;
  float v = acc[i] + b[i % HC];
  if (relu) v = v > 0.f ? v : 0.f;
  out[i] = v;
}

// ---------------------------------------------------------------- layer driver
static void run_layer(const float* xin, int K, const float* W, const float* a_s,
                      const float* a_d, const float* bias, int H, int C, int relu,
                      const long long* ei, _Float16* xh, _Float16* wh, float* hbuf,
                      float* als, float* ald, float* emax, float* denom,
                      float* accout, hipStream_t stream) {
  const int HC = H * C;
  const int nx = N_NODES * K;
  k_f32_to_f16<<<(nx + 255) / 256, 256, 0, stream>>>(xin, xh, nx);
  k_w_transpose_f16<<<(K * HC + 255) / 256, 256, 0, stream>>>(W, wh, K, HC);
  const int mtiles = N_NODES / 16;
  if (HC % 64 == 0) {
    const int total = mtiles * (HC / 64);
    k_wmma_gemm<4><<<(total + 7) / 8, 256, 0, stream>>>(xh, wh, hbuf, N_NODES, HC, K);
  } else {
    const int total = mtiles * (HC / 32);
    k_wmma_gemm<2><<<(total + 7) / 8, 256, 0, stream>>>(xh, wh, hbuf, N_NODES, HC, K);
  }
  k_att_init<<<(N_NODES * H + 255) / 256, 256, 0, stream>>>(hbuf, a_s, a_d, als, ald,
                                                            emax, denom, H, C);
  k_zero_f32<<<(N_NODES * HC + 255) / 256, 256, 0, stream>>>(accout, N_NODES * HC);
  const int ne = E_TOTAL * H;
  k_edge_max<<<(ne + 255) / 256, 256, 0, stream>>>(ei, als, ald, emax, H);
  k_edge_denom<<<(ne + 255) / 256, 256, 0, stream>>>(ei, als, ald, emax, denom, H);
  k_edge_msg<<<(E_TOTAL + 7) / 8, 256, 0, stream>>>(ei, als, ald, emax, denom, hbuf,
                                                    accout, H, C);
  k_bias_act<<<(N_NODES * HC + 255) / 256, 256, 0, stream>>>(accout, bias, accout,
                                                             HC, relu);
}

extern "C" void kernel_launch(void* const* d_in, const int* in_sizes, int n_in,
                              void* d_out, int out_size, void* d_ws, size_t ws_size,
                              hipStream_t stream) {
  const float*     x  = (const float*)d_in[0];
  const long long* ei = (const long long*)d_in[1];
  const float *W1 = (const float*)d_in[2],  *as1 = (const float*)d_in[3],
              *ad1 = (const float*)d_in[4], *b1  = (const float*)d_in[5];
  const float *W2 = (const float*)d_in[6],  *as2 = (const float*)d_in[7],
              *ad2 = (const float*)d_in[8], *b2  = (const float*)d_in[9];
  const float *W3 = (const float*)d_in[10], *as3 = (const float*)d_in[11],
              *ad3 = (const float*)d_in[12], *b3 = (const float*)d_in[13];
  const float *W4 = (const float*)d_in[14], *as4 = (const float*)d_in[15],
              *ad4 = (const float*)d_in[16], *b4 = (const float*)d_in[17];

  // ---- workspace carve (256B aligned), total ~141 MB
  char* p = (char*)d_ws;
  auto take = [&](size_t bytes) {
    char* r = p; p += (bytes + 255) & ~size_t(255); return (void*)r;
  };
  float*    actA  = (float*)take((size_t)N_NODES * 256 * 4);   // layer 1/3 outputs
  float*    actB  = (float*)take((size_t)N_NODES * 32  * 4);   // layer 2 output
  float*    hbuf  = (float*)take((size_t)N_NODES * 256 * 4);   // per-layer h = xW
  _Float16* xh    = (_Float16*)take((size_t)N_NODES * 256 * 2);
  _Float16* wh    = (_Float16*)take((size_t)256 * 256 * 2);
  float*    als   = (float*)take((size_t)N_NODES * 8 * 4);
  float*    ald   = (float*)take((size_t)N_NODES * 8 * 4);
  float*    emax  = (float*)take((size_t)N_NODES * 8 * 4);
  float*    denom = (float*)take((size_t)N_NODES * 8 * 4);
  float*    out   = (float*)d_out;                              // [N, 128]

  // encoder
  run_layer(x,    128, W1, as1, ad1, b1, 8, 32,  1, ei, xh, wh, hbuf,
            als, ald, emax, denom, actA, stream);               // relu
  run_layer(actA, 256, W2, as2, ad2, b2, 1, 32,  0, ei, xh, wh, hbuf,
            als, ald, emax, denom, actB, stream);
  // decoder
  run_layer(actB, 32,  W3, as3, ad3, b3, 8, 32,  1, ei, xh, wh, hbuf,
            als, ald, emax, denom, actA, stream);               // relu
  run_layer(actA, 256, W4, as4, ad4, b4, 1, 128, 0, ei, xh, wh, hbuf,
            als, ald, emax, denom, out, stream);
}
